// EntityAttentionLayer_21741124452848
// MI455X (gfx1250) — matrix operand order repaired and checked by reading, not compiled, not run on hardware
//
#include <hip/hip_runtime.h>

typedef __attribute__((ext_vector_type(16))) __bf16 v16bf;
typedef __attribute__((ext_vector_type(8)))  float  v8f;

#define DEV_INLINE __device__ __forceinline__

constexpr int B_ = 8, T_ = 2048, H_ = 768, M_ = 4096, E_ = 32000, D_ = 256;
constexpr int K2H = 1536; // 2*H
constexpr float SMALL_ = 1e-8f;
constexpr float LN_EPS = 1e-12f;

union BFrag {
  v16bf v;
  uint4 q[2];
  __bf16 e[16];
};

DEV_INLINE __bf16 f2bf(float f) {
  unsigned u = __float_as_uint(f);
  unsigned r = (u + 0x7FFFu + ((u >> 16) & 1u)) >> 16;
  unsigned short s = (unsigned short)r;
  return __builtin_bit_cast(__bf16, s);
}

// ---------------------------------------------------------------- copy input
__global__ __launch_bounds__(256) void copy_f4(const float4* __restrict__ in,
                                               float4* __restrict__ out, int n4) {
  int i = blockIdx.x * 256 + threadIdx.x;
  if (i < n4) out[i] = in[i];
}

// ------------------------------------------- fp32 -> bf16 (row-major + transposed)
__global__ __launch_bounds__(256) void conv_transpose_bf16(
    const float* __restrict__ in, __bf16* __restrict__ out_rm,
    __bf16* __restrict__ out_tr, int R, int C) {
  __shared__ __bf16 tile[32][33];
  int tx = threadIdx.x, ty = threadIdx.y;
  int c0 = blockIdx.x * 32, r0 = blockIdx.y * 32;
#pragma unroll
  for (int j = 0; j < 4; ++j) {
    int y = ty + j * 8;
    float v = in[(size_t)(r0 + y) * C + c0 + tx];
    __bf16 bv = f2bf(v);
    tile[y][tx] = bv;
    if (out_rm) out_rm[(size_t)(r0 + y) * C + c0 + tx] = bv;
  }
  __syncthreads();
#pragma unroll
  for (int j = 0; j < 4; ++j) {
    int y = ty + j * 8;
    out_tr[(size_t)(c0 + y) * R + r0 + tx] = tile[tx][y];
  }
}

// ---------------------------------------------------------------- entity norms
__global__ __launch_bounds__(256) void entity_norms(const float* __restrict__ ent,
                                                    float* __restrict__ enorm) {
  int wave = threadIdx.x >> 5, lane = threadIdx.x & 31;
  int row = blockIdx.x * 8 + wave;
  const float* p = ent + (size_t)row * D_;
  float s = 0.f;
#pragma unroll
  for (int i = 0; i < 8; ++i) { float v = p[lane + i * 32]; s += v * v; }
#pragma unroll
  for (int off = 16; off > 0; off >>= 1) s += __shfl_xor(s, off, 32);
  if (lane == 0) enorm[row] = sqrtf(s);
}

// --------------------- gather mention rows, Q = concat(start,end)@wq + bq (WMMA)
__global__ __launch_bounds__(256) void mention_proj(
    const float* __restrict__ enc, const int* __restrict__ mbp,
    const int* __restrict__ msp, const int* __restrict__ mep,
    const __bf16* __restrict__ wqT, const float* __restrict__ bq,
    float* __restrict__ q_out, __bf16* __restrict__ q_bf,
    float* __restrict__ mnorm) {
  __shared__ __align__(16) __bf16 a_lds[16 * K2H];
  __shared__ float msq[16];
  __shared__ int srow[16], erow[16];
  int tid = threadIdx.x;
  int mbase = blockIdx.x * 16;
  if (tid < 16) {
    int m = mbase + tid;
    int b = mbp[m];
    srow[tid] = (b * T_ + msp[m]) * H_;
    erow[tid] = (b * T_ + mep[m]) * H_;
    msq[tid] = 0.f;
  }
  __syncthreads();
  for (int idx = tid; idx < 16 * K2H; idx += 256) {
    int m = idx / K2H, k = idx % K2H;
    float v = (k < H_) ? enc[srow[m] + k] : enc[erow[m] + k - H_];
    a_lds[idx] = f2bf(v);
  }
  __syncthreads();
  int wave = tid >> 5, lane = tid & 31;
  int row = lane & 15, hi = lane >> 4;
  int n0 = wave * 32;
  v8f c0v = {}; v8f c1v = {};
  for (int kk = 0; kk < K2H / 32; ++kk) {
    int kb = kk * 32;
    BFrag a, b0, b1;
    const __bf16* ap = &a_lds[row * K2H + kb + hi * 16];
    a.q[0] = *reinterpret_cast<const uint4*>(ap);
    a.q[1] = *reinterpret_cast<const uint4*>(ap + 8);
    const __bf16* bp0 = wqT + (size_t)(n0 + row) * K2H + kb + hi * 16;
    b0.q[0] = *reinterpret_cast<const uint4*>(bp0);
    b0.q[1] = *reinterpret_cast<const uint4*>(bp0 + 8);
    const __bf16* bp1 = wqT + (size_t)(n0 + 16 + row) * K2H + kb + hi * 16;
    b1.q[0] = *reinterpret_cast<const uint4*>(bp1);
    b1.q[1] = *reinterpret_cast<const uint4*>(bp1 + 8);
    c0v = __builtin_amdgcn_wmma_f32_16x16x32_bf16(false, a.v, false, b0.v,
                                                  (short)0, c0v, false, false);
    c1v = __builtin_amdgcn_wmma_f32_16x16x32_bf16(false, a.v, false, b1.v,
                                                  (short)0, c1v, false, false);
  }
#pragma unroll
  for (int t = 0; t < 2; ++t) {
    v8f c = t ? c1v : c0v;
    int n = n0 + t * 16 + (lane & 15);
    float bias = bq[n];
#pragma unroll
    for (int r = 0; r < 8; ++r) {
      int m = r + hi * 8;
      float val = c[r] + bias;
      q_out[(size_t)(mbase + m) * D_ + n] = val;
      q_bf[(size_t)(mbase + m) * D_ + n] = f2bf(val);
      atomicAdd(&msq[m], val * val);
    }
  }
  __syncthreads();
  if (tid < 16) mnorm[mbase + tid] = sqrtf(msq[tid]);
}

// ----------------------- scores = Q @ Eᵀ (WMMA), write raw scores + cos_scores
__global__ __launch_bounds__(256) void scores_kernel(
    const __bf16* __restrict__ q_bf, const __bf16* __restrict__ ent_bf,
    const float* __restrict__ mnorm, const float* __restrict__ enorm,
    float* __restrict__ raw, float* __restrict__ cosb) {
  int tid = threadIdx.x;
  int wave = tid >> 5, lane = tid & 31;
  int row = lane & 15, hi = lane >> 4;
  int mbase = blockIdx.y * 16;
  int e = blockIdx.x * 128 + wave * 16 + row;
  v8f c = {};
#pragma unroll
  for (int kk = 0; kk < D_ / 32; ++kk) {
    int kb = kk * 32;
    BFrag a, b;
    const __bf16* ap = q_bf + (size_t)(mbase + row) * D_ + kb + hi * 16;
    a.q[0] = *reinterpret_cast<const uint4*>(ap);
    a.q[1] = *reinterpret_cast<const uint4*>(ap + 8);
    const __bf16* bp = ent_bf + (size_t)e * D_ + kb + hi * 16;
    b.q[0] = *reinterpret_cast<const uint4*>(bp);
    b.q[1] = *reinterpret_cast<const uint4*>(bp + 8);
    c = __builtin_amdgcn_wmma_f32_16x16x32_bf16(false, a.v, false, b.v,
                                                (short)0, c, false, false);
  }
  float ie = 1.0f / (SMALL_ + enorm[e]);
#pragma unroll
  for (int r = 0; r < 8; ++r) {
    int m = mbase + r + hi * 8;
    float s = c[r];
    size_t o = (size_t)m * E_ + e;
    raw[o] = s;
    cosb[o] = s * ie / (SMALL_ + mnorm[m]);
  }
}

// ---------------------------------------- per-row online softmax max/sum
__global__ __launch_bounds__(256) void softmax_stats(const float* __restrict__ raw,
                                                     float* __restrict__ rmax,
                                                     float* __restrict__ rinv) {
  __shared__ float smx[256], ssm[256];
  int m = blockIdx.x, tid = threadIdx.x;
  const float* p = raw + (size_t)m * E_;
  float mx = -3.4e38f, sm = 0.f;
  for (int i = tid; i < E_; i += 256) {
    float v = p[i];
    if (v > mx) { sm = sm * __expf(mx - v) + 1.0f; mx = v; }
    else sm += __expf(v - mx);
  }
  smx[tid] = mx; ssm[tid] = sm;
  __syncthreads();
  for (int s = 128; s > 0; s >>= 1) {
    if (tid < s) {
      float m1 = smx[tid], s1 = ssm[tid];
      float m2 = smx[tid + s], s2 = ssm[tid + s];
      float nm = fmaxf(m1, m2);
      smx[tid] = nm;
      ssm[tid] = s1 * __expf(m1 - nm) + s2 * __expf(m2 - nm);
    }
    __syncthreads();
  }
  if (tid == 0) { rmax[m] = smx[0]; rinv[m] = 1.0f / ssm[0]; }
}

// ------- normalize weights in place + retrieved = W @ E (WMMA, K-split by wave)
__global__ __launch_bounds__(256) void attn_retrieve(
    float* __restrict__ att, const __bf16* __restrict__ entT,
    const float* __restrict__ rmax, const float* __restrict__ rinv,
    float* __restrict__ retr) {
  __shared__ float acc[16 * D_];
  int tid = threadIdx.x;
  int wave = tid >> 5, lane = tid & 31;
  int row = lane & 15, hi = lane >> 4;
  int mbase = blockIdx.x * 16;
  int m = mbase + row;
  float mx = rmax[m], iv = rinv[m];
  float* wrow = att + (size_t)m * E_;
  v8f czero = {};
  v8f c[16];
#pragma unroll
  for (int t = 0; t < 16; ++t) c[t] = czero;

  for (int kc = wave; kc < E_ / 32; kc += 8) {
    int kb = kc * 32;
    float sv[16];
    {
      const float4* p = reinterpret_cast<const float4*>(wrow + kb + hi * 16);
      float4 x0 = p[0], x1 = p[1];
      const float4* q = reinterpret_cast<const float4*>(wrow + kb + 8 + hi * 16);
      float4 y0 = q[0], y1 = q[1];
      sv[0] = x0.x; sv[1] = x0.y; sv[2] = x0.z; sv[3] = x0.w;
      sv[4] = x1.x; sv[5] = x1.y; sv[6] = x1.z; sv[7] = x1.w;
      sv[8] = y0.x; sv[9] = y0.y; sv[10] = y0.z; sv[11] = y0.w;
      sv[12] = y1.x; sv[13] = y1.y; sv[14] = y1.z; sv[15] = y1.w;
    }
    BFrag a;
    float wv[16];
#pragma unroll
    for (int j = 0; j < 16; ++j) {
      float w = __expf(sv[j] - mx) * iv;
      wv[j] = w;
      a.e[j] = f2bf(w);
    }
    {
      float4* p = reinterpret_cast<float4*>(wrow + kb + hi * 16);
      p[0] = make_float4(wv[0], wv[1], wv[2], wv[3]);
      p[1] = make_float4(wv[4], wv[5], wv[6], wv[7]);
      float4* q = reinterpret_cast<float4*>(wrow + kb + 8 + hi * 16);
      q[0] = make_float4(wv[8], wv[9], wv[10], wv[11]);
      q[1] = make_float4(wv[12], wv[13], wv[14], wv[15]);
    }
    int col = lane & 15;
#pragma unroll
    for (int t = 0; t < 16; ++t) {
      BFrag b;
      const __bf16* bp = entT + (size_t)(t * 16 + col) * E_ + kb + hi * 16;
      b.q[0] = *reinterpret_cast<const uint4*>(bp);
      b.q[1] = *reinterpret_cast<const uint4*>(bp + 8);
      c[t] = __builtin_amdgcn_wmma_f32_16x16x32_bf16(false, a.v, false, b.v,
                                                     (short)0, c[t], false, false);
    }
  }
  // cross-wave K reduction through LDS
  for (int g = 0; g < 8; ++g) {
    if (wave == g) {
#pragma unroll
      for (int t = 0; t < 16; ++t) {
        int n = t * 16 + (lane & 15);
#pragma unroll
        for (int r = 0; r < 8; ++r) {
          int ml = r + hi * 8;
          if (g == 0) acc[ml * D_ + n] = c[t][r];
          else acc[ml * D_ + n] += c[t][r];
        }
      }
    }
    __syncthreads();
  }
  for (int i = tid; i < 16 * D_; i += 256)
    retr[(size_t)mbase * D_ + i] = acc[i];
}

// ------------------------- upd = (retrieved @ we + be)*mask, scatter-add
__global__ __launch_bounds__(256) void project_scatter(
    const float* __restrict__ retr, const float* __restrict__ we,
    const float* __restrict__ be, const float* __restrict__ mask,
    const int* __restrict__ mbp, const int* __restrict__ msp,
    float* __restrict__ enc_out) {
  __shared__ float r[D_];
  int m = blockIdx.x, tid = threadIdx.x;
  r[tid] = retr[(size_t)m * D_ + tid];
  __syncthreads();
  int b = mbp[m], s = msp[m];
  float mk = mask[m];
  size_t base = ((size_t)b * T_ + s) * H_;
#pragma unroll
  for (int j = 0; j < 3; ++j) {
    int h = tid + j * 256;
    float acc = 0.f;
    for (int k = 0; k < D_; ++k) acc += r[k] * we[k * H_ + h];
    atomicAdd(&enc_out[base + h], (acc + be[h]) * mk);
  }
}

// ---------------------------------------------------- in-place LayerNorm
__global__ __launch_bounds__(256) void layernorm_inplace(
    float* __restrict__ enc, const float* __restrict__ scale,
    const float* __restrict__ bias) {
  __shared__ float red[256];
  size_t base = (size_t)blockIdx.x * H_;
  int tid = threadIdx.x;
  float v0 = enc[base + tid], v1 = enc[base + tid + 256], v2 = enc[base + tid + 512];
  red[tid] = v0 + v1 + v2;
  __syncthreads();
  for (int s = 128; s > 0; s >>= 1) { if (tid < s) red[tid] += red[tid + s]; __syncthreads(); }
  float mu = red[0] / (float)H_;
  __syncthreads();
  float d0 = v0 - mu, d1 = v1 - mu, d2 = v2 - mu;
  red[tid] = d0 * d0 + d1 * d1 + d2 * d2;
  __syncthreads();
  for (int s = 128; s > 0; s >>= 1) { if (tid < s) red[tid] += red[tid + s]; __syncthreads(); }
  float rs = rsqrtf(red[0] / (float)H_ + LN_EPS);
  enc[base + tid] = d0 * rs * scale[tid] + bias[tid];
  enc[base + tid + 256] = d1 * rs * scale[tid + 256] + bias[tid + 256];
  enc[base + tid + 512] = d2 * rs * scale[tid + 512] + bias[tid + 512];
}

extern "C" void kernel_launch(void* const* d_in, const int* in_sizes, int n_in,
                              void* d_out, int out_size, void* d_ws, size_t ws_size,
                              hipStream_t stream) {
  const float* enc_in = (const float*)d_in[0];
  const int* mbp = (const int*)d_in[1];
  const int* msp = (const int*)d_in[2];
  const int* mep = (const int*)d_in[3];
  const float* mask = (const float*)d_in[4];
  const float* ent = (const float*)d_in[5];
  const float* wq = (const float*)d_in[6];
  const float* bq = (const float*)d_in[7];
  const float* we = (const float*)d_in[8];
  const float* be = (const float*)d_in[9];
  const float* ln_s = (const float*)d_in[10];
  const float* ln_b = (const float*)d_in[11];

  float* out = (float*)d_out;
  float* out_enc = out;                                  // [B,T,H]
  float* out_q = out + (size_t)B_ * T_ * H_;             // [M,D]
  float* out_cos = out_q + (size_t)M_ * D_;              // [M,E]
  float* out_att = out_cos + (size_t)M_ * E_;            // [M,E]

  char* ws = (char*)d_ws;
  __bf16* ent_bf  = (__bf16*)(ws);                       // E*D bf16 (row-major)
  __bf16* entT_bf = (__bf16*)(ws + 16384000);            // D*E bf16 (transposed)
  __bf16* wqT_bf  = (__bf16*)(ws + 32768000);            // D x 2H bf16 (transposed)
  __bf16* q_bf    = (__bf16*)(ws + 33554432);            // M*D bf16
  float* e_nrm    = (float*)(ws + 35651584);             // E
  float* m_nrm    = (float*)(ws + 35779584);             // M
  float* r_max    = (float*)(ws + 35795968);             // M
  float* r_inv    = (float*)(ws + 35812352);             // M
  float* retr     = (float*)(ws + 35828736);             // M*D fp32

  (void)in_sizes; (void)n_in; (void)out_size; (void)ws_size;

  copy_f4<<<dim3((B_ * T_ * H_) / 4 / 256), 256, 0, stream>>>(
      (const float4*)enc_in, (float4*)out_enc, (B_ * T_ * H_) / 4);
  conv_transpose_bf16<<<dim3(D_ / 32, E_ / 32), dim3(32, 8), 0, stream>>>(
      ent, ent_bf, entT_bf, E_, D_);
  conv_transpose_bf16<<<dim3(D_ / 32, K2H / 32), dim3(32, 8), 0, stream>>>(
      wq, nullptr, wqT_bf, K2H, D_);
  entity_norms<<<E_ / 8, 256, 0, stream>>>(ent, e_nrm);
  mention_proj<<<M_ / 16, 256, 0, stream>>>(enc_in, mbp, msp, mep, wqT_bf, bq,
                                            out_q, q_bf, m_nrm);
  scores_kernel<<<dim3(E_ / 128, M_ / 16), 256, 0, stream>>>(
      q_bf, ent_bf, m_nrm, e_nrm, out_att, out_cos);
  softmax_stats<<<M_, 256, 0, stream>>>(out_att, r_max, r_inv);
  attn_retrieve<<<M_ / 16, 256, 0, stream>>>(out_att, entT_bf, r_max, r_inv, retr);
  project_scatter<<<M_, 256, 0, stream>>>(retr, we, be, mask, mbp, msp, out_enc);
  layernorm_inplace<<<B_ * T_, 256, 0, stream>>>(out_enc, ln_s, ln_b);
}
